// Vanilla_DisGNN_67534065762903
// MI455X (gfx1250) — compile-verified, matrix operand order checked by reading, not run on hardware
//
#include <hip/hip_runtime.h>

#define HIDDEN 128
#define NGRAPHS 64

typedef __attribute__((ext_vector_type(16))) __bf16 v16bf;
typedef __attribute__((ext_vector_type(8)))  float  v8f;

// ---------- helpers ----------

// cheap f32 -> bf16 (round-half-up): 2 VALU ops
__device__ __forceinline__ unsigned short f2bfu(float f) {
    union { float f; unsigned u; } v; v.f = f;
    return (unsigned short)((v.u + 0x8000u) >> 16);
}

// two f32 -> packed 2x bf16 in one dword: 2 adds + v_perm_b32
__device__ __forceinline__ unsigned pack_bf16(float lo, float hi) {
    union { float f; unsigned u; } a, b;
    a.f = lo; b.f = hi;
    // bytes: [3,2]=hi[3,2], [1,0]=lo[3,2]  (sel 0-3 -> src1, 4-7 -> src0)
    return __builtin_amdgcn_perm(b.u + 0x8000u, a.u + 0x8000u, 0x07060302u);
}

__device__ __forceinline__ __bf16 u2bf(unsigned short s) {
    union { unsigned short s; __bf16 b; } c; c.s = s; return c.b;
}

__device__ __forceinline__ float silu(float x) {
    return x * (1.0f / (1.0f + __expf(-x)));
}

// A fragment: 16x32 bf16 tile from LDS (row-major, stride HIDDEN ushorts).
// ISA 7.12.2: lane L holds row M=L&15; kbase = 8*(L>>4); VGPR v<4 -> K=kb+2v,
// VGPR v>=4 -> K=16+kb+2(v-4). Pairs are contiguous -> lowers to 2x ds_load_b128.
__device__ __forceinline__ v16bf load_a(const unsigned short* s, int mrow, int koff, int lane) {
    const int m  = lane & 15;
    const int kb = koff + ((lane >> 4) << 3);
    const unsigned short* rp = s + (mrow + m) * HIDDEN;
    v16bf a;
#pragma unroll
    for (int v = 0; v < 8; ++v) {
        int k0 = kb + ((v < 4) ? (2 * v) : (16 + 2 * (v - 4)));
        a[2 * v]     = u2bf(rp[k0]);
        a[2 * v + 1] = u2bf(rp[k0 + 1]);
    }
    return a;
}

// B fragment from *fragment-major* LDS buffer: each (frag, lane) slot is 16
// contiguous bf16 (32 bytes, 32B-aligned) -> exactly 2x ds_load_b128.
__device__ __forceinline__ v16bf load_bfrag(const unsigned short* w, int frag, int lane) {
    return *(const v16bf*)(w + ((frag * 32 + lane) << 4));
}

// Cooperative: stage a 128x128 f32 weight matrix (global, K-major [k][n]) into
// fragment-major bf16 LDS. frag = kt*8 + j. Each thread owns whole slots:
// gathers 16 f32 (coalesced across nlo lanes), packs, stores 2x b128.
__device__ __forceinline__ void stage_weights(unsigned short* wbuf,
                                              const float* __restrict__ W, int tid) {
    for (int slot = tid; slot < 4 * 8 * 32; slot += 256) {
        int lane = slot & 31;
        int frag = slot >> 5;                 // kt*8 + j
        int kt = frag >> 3, j = frag & 7;
        int nlo = lane & 15, lanehi = lane >> 4;
        int n  = (j << 4) | nlo;
        int kb = (kt << 5) + (lanehi << 3);
        unsigned packed[8];
#pragma unroll
        for (int v = 0; v < 8; ++v) {
            int k0 = kb + ((v < 4) ? (2 * v) : (16 + 2 * (v - 4)));
            packed[v] = pack_bf16(W[k0 * HIDDEN + n], W[(k0 + 1) * HIDDEN + n]);
        }
        unsigned* dst = (unsigned*)(wbuf + (slot << 4));
#pragma unroll
        for (int v = 0; v < 8; ++v) dst[v] = packed[v];
    }
}

// Same for e_lin [16][128] with K zero-padded to 32; only frags j=0..7 (kt=0).
__device__ __forceinline__ void stage_elin(unsigned short* wbuf,
                                           const float* __restrict__ W, int tid) {
    for (int slot = tid; slot < 8 * 32; slot += 256) {
        int lane = slot & 31;
        int j = slot >> 5;
        int nlo = lane & 15, lanehi = lane >> 4;
        int n  = (j << 4) | nlo;
        int kb = lanehi << 3;
        unsigned packed[8];
#pragma unroll
        for (int v = 0; v < 4; ++v) {       // K = kb..kb+7 (real, < 16)
            int k0 = kb + 2 * v;
            packed[v] = pack_bf16(W[k0 * HIDDEN + n], W[(k0 + 1) * HIDDEN + n]);
        }
#pragma unroll
        for (int v = 4; v < 8; ++v) packed[v] = 0;   // K >= 16: zero pad
        unsigned* dst = (unsigned*)(wbuf + (slot << 4));
#pragma unroll
        for (int v = 0; v < 8; ++v) dst[v] = packed[v];
    }
}

// ---------- residual MLP block: y = x + silu(silu(x@W0+b0)@W1+b1) ----------
// W: [2][128][128] (in,out), B2: [2][128]. 128 rows per block, 8 waves.
__global__ __launch_bounds__(256) void residual_kernel(
    const float* __restrict__ x, const float* __restrict__ W,
    const float* __restrict__ B2, float* __restrict__ y, int nrows)
{
    __shared__ __align__(32) unsigned short stage[HIDDEN * HIDDEN]; // 32 KB: x, then h (bf16)
    __shared__ __align__(32) unsigned short wbuf[HIDDEN * HIDDEN];  // 32 KB: W0/W1 frag-major

    const int tid  = threadIdx.x;
    const int lane = tid & 31;
    const int wave = tid >> 5;
    const int rowBase = blockIdx.x * 128;

    // stage x tile (zero-fill OOB rows): float4 loads, packed b64 LDS stores
    for (int idx = tid; idx < (HIDDEN * HIDDEN) / 4; idx += 256) {
        int r = idx >> 5, c4 = (idx & 31) << 2;
        int gr = rowBase + r;
        float4 v = make_float4(0.0f, 0.0f, 0.0f, 0.0f);
        if (gr < nrows) v = *(const float4*)(x + (long)gr * HIDDEN + c4);
        unsigned* dst = (unsigned*)(stage + r * HIDDEN + c4);
        dst[0] = pack_bf16(v.x, v.y);
        dst[1] = pack_bf16(v.z, v.w);
    }
    stage_weights(wbuf, W, tid);
    __syncthreads();

    const int mrow = wave * 16;
    const int nlo  = lane & 15;
    const int mhi  = (lane >> 4) << 3;

    // ---- GEMM1: h = silu(x @ W0 + b0) ----
    v16bf A[4];
#pragma unroll
    for (int kt = 0; kt < 4; ++kt) A[kt] = load_a(stage, mrow, kt * 32, lane);

#pragma unroll
    for (int j = 0; j < 8; ++j) {
        v8f c = {};
#pragma unroll
        for (int kt = 0; kt < 4; ++kt) {
            v16bf b = load_bfrag(wbuf, kt * 8 + j, lane);
            c = __builtin_amdgcn_wmma_f32_16x16x32_bf16(false, A[kt], false, b,
                                                        (short)0, c, false, false);
        }
        const int n = (j << 4) + nlo;
        const float bn = B2[n];
#pragma unroll
        for (int r = 0; r < 8; ++r) {
            float h = silu(c[r] + bn);
            int m = mrow + r + mhi;            // wave-local row: no cross-wave race
            stage[m * HIDDEN + n] = f2bfu(h);
        }
    }
    __syncthreads();
    stage_weights(wbuf, W + HIDDEN * HIDDEN, tid);
    __syncthreads();

    // ---- GEMM2: y = x + silu(h @ W1 + b1) ----
#pragma unroll
    for (int kt = 0; kt < 4; ++kt) A[kt] = load_a(stage, mrow, kt * 32, lane);

#pragma unroll
    for (int j = 0; j < 8; ++j) {
        v8f c = {};
#pragma unroll
        for (int kt = 0; kt < 4; ++kt) {
            v16bf b = load_bfrag(wbuf, kt * 8 + j, lane);
            c = __builtin_amdgcn_wmma_f32_16x16x32_bf16(false, A[kt], false, b,
                                                        (short)0, c, false, false);
        }
        const int n = (j << 4) + nlo;
        const float bn = B2[HIDDEN + n];
#pragma unroll
        for (int r = 0; r < 8; ++r) {
            int gm = rowBase + mrow + r + mhi;
            if (gm < nrows) {
                float h = silu(c[r] + bn);
                long o = (long)gm * HIDDEN + n;
                y[o] = x[o] + h;               // in-place safe (same lane r/w same elem)
            }
        }
    }
}

// ---------- fused edge kernel ----------
// filter = (ef @ e_lin) * smooth; val = filter * s_dst[col]; atomicAdd conv[row]
// 8 waves x 16 edges = 128 edges per block.
__global__ __launch_bounds__(256) void edge_kernel(
    const float* __restrict__ ef, const float* __restrict__ elin,
    const float* __restrict__ smooth, const int* __restrict__ erow,
    const int* __restrict__ ecol, const float* __restrict__ sdst,
    float* __restrict__ conv, int E)
{
    __shared__ __align__(32) unsigned short elbuf[8 * 32 * 16];  // 8 KB frag-major e_lin

    const int tid  = threadIdx.x;
    const int lane = tid & 31;
    const int wave = tid >> 5;

    stage_elin(elbuf, elin, tid);
    __syncthreads();

    const long ebase = (long)blockIdx.x * 128 + wave * 16;

    // A fragment: lane owns edge m = lane&15; real K only 0..15 (upper half zero)
    const int kb = (lane >> 4) << 3;
    const long ea = ebase + (lane & 15);
    v16bf a;
#pragma unroll
    for (int v = 0; v < 8; ++v) { a[2 * v] = u2bf(0); a[2 * v + 1] = u2bf(0); }
    if (ea < E) {
        const float* rp = ef + ea * 16;
#pragma unroll
        for (int v = 0; v < 4; ++v) {
            a[2 * v]     = u2bf(f2bfu(rp[kb + 2 * v]));
            a[2 * v + 1] = u2bf(f2bfu(rp[kb + 2 * v + 1]));
        }
    }

    // per-lane metadata for the 8 output edges this lane holds in C tiles
    const int mhi = (lane >> 4) << 3;
    int   rowi[8], coli[8];
    float sm[8];
    bool  valid[8];
#pragma unroll
    for (int r = 0; r < 8; ++r) {
        long e = ebase + r + mhi;
        valid[r] = (e < E);
        rowi[r] = valid[r] ? erow[e] : 0;
        coli[r] = valid[r] ? ecol[e] : 0;
        sm[r]   = valid[r] ? smooth[e] : 0.0f;
    }

    const int nlo = lane & 15;
#pragma unroll
    for (int j = 0; j < 8; ++j) {
        v8f c = {};
        v16bf b = load_bfrag(elbuf, j, lane);
        c = __builtin_amdgcn_wmma_f32_16x16x32_bf16(false, a, false, b,
                                                    (short)0, c, false, false);
        const int n = (j << 4) + nlo;
#pragma unroll
        for (int r = 0; r < 8; ++r) {
            if (valid[r]) {
                float val = c[r] * sm[r] * sdst[(long)coli[r] * HIDDEN + n];
                atomicAdd(&conv[(long)rowi[r] * HIDDEN + n], val);
            }
        }
    }
}

// ---------- elementwise kernels (float4-vectorized; NH % 4 == 0) ----------

__global__ void rowscale_kernel(float4* __restrict__ a, const float* __restrict__ Cv, long total4) {
    long i = (long)blockIdx.x * blockDim.x + threadIdx.x;
    if (i < total4) {
        float s = Cv[i >> 5];              // 32 float4 per row of 128
        float4 v = a[i];
        v.x *= s; v.y *= s; v.z *= s; v.w *= s;
        a[i] = v;
    }
}

__global__ void mul_kernel(float4* __restrict__ a, const float4* __restrict__ b, long total4) {
    long i = (long)blockIdx.x * blockDim.x + threadIdx.x;
    if (i < total4) {
        float4 u = a[i], v = b[i];
        u.x *= v.x; u.y *= v.y; u.z *= v.z; u.w *= v.w;
        a[i] = u;
    }
}

__global__ void add_kernel(float4* __restrict__ a, const float4* __restrict__ b, long total4) {
    long i = (long)blockIdx.x * blockDim.x + threadIdx.x;
    if (i < total4) {
        float4 u = a[i], v = b[i];
        u.x += v.x; u.y += v.y; u.z += v.z; u.w += v.w;
        a[i] = u;
    }
}

// ---------- graph pooling: segment_sum over sorted batch_index ----------
__global__ __launch_bounds__(256) void pool_kernel(
    const float* __restrict__ s, const int* __restrict__ batch,
    float* __restrict__ out, int n)
{
    __shared__ float acc[NGRAPHS * HIDDEN];   // 32 KB
    const int tid = threadIdx.x;
    for (int i = tid; i < NGRAPHS * HIDDEN; i += 256) acc[i] = 0.0f;
    __syncthreads();

    const long base = (long)blockIdx.x * 512;
    for (long idx = tid; idx < 512L * HIDDEN; idx += 256) {
        long node = base + (idx >> 7);
        if (node < n) {
            int g = batch[node];
            int c = (int)(idx & 127);
            atomicAdd(&acc[g * HIDDEN + c], s[node * HIDDEN + c]);
        }
    }
    __syncthreads();
    for (int i = tid; i < NGRAPHS * HIDDEN; i += 256) {
        float v = acc[i];
        if (v != 0.0f) atomicAdd(&out[i], v);
    }
}

// ---------- host ----------

extern "C" void kernel_launch(void* const* d_in, const int* in_sizes, int n_in,
                              void* d_out, int out_size, void* d_ws, size_t ws_size,
                              hipStream_t stream) {
    const float* scalar      = (const float*)d_in[0];
    const float* ef          = (const float*)d_in[1];
    const int*   edge_index  = (const int*)d_in[2];
    const float* Cv          = (const float*)d_in[3];
    const float* conv_smooth = (const float*)d_in[4];
    const int*   batch       = (const int*)d_in[5];
    const float* res_W       = (const float*)d_in[6];  // [15][2][128][128]
    const float* res_b       = (const float*)d_in[7];  // [15][2][128]
    const float* e_lin       = (const float*)d_in[8];  // [2][16][128]
    float* out = (float*)d_out;

    const int N = in_sizes[0] / HIDDEN;     // 100000
    const int E = in_sizes[2] / 2;          // 1600000
    const long NH = (long)N * HIDDEN;
    const long NH4 = NH / 4;

    float* s_cur = (float*)d_ws;
    float* s_src = s_cur + NH;
    float* s_dst = s_src + NH;
    float* conv  = s_dst + NH;

    const int resBlocks  = (N + 127) / 128;
    const int edgeBlocks = (E + 127) / 128;
    const int ewBlocks   = (int)((NH4 + 255) / 256);
    const int poolBlocks = (N + 511) / 512;

    auto Wp = [&](int k) { return res_W + (long)k * 2 * HIDDEN * HIDDEN; };
    auto bp = [&](int k) { return res_b + (long)k * 2 * HIDDEN; };

    hipMemcpyAsync(s_cur, scalar, NH * sizeof(float), hipMemcpyDeviceToDevice, stream);

    for (int i = 0; i < 2; ++i) {
        const int p = 6 * i;
        residual_kernel<<<resBlocks, 256, 0, stream>>>(s_cur, Wp(p),     bp(p),     s_src, N);
        residual_kernel<<<resBlocks, 256, 0, stream>>>(s_cur, Wp(p + 1), bp(p + 1), s_dst, N);

        hipMemsetAsync(conv, 0, NH * sizeof(float), stream);
        edge_kernel<<<edgeBlocks, 256, 0, stream>>>(ef, e_lin + (long)i * 16 * HIDDEN,
                                                    conv_smooth, edge_index, edge_index + E,
                                                    s_dst, conv, E);
        rowscale_kernel<<<ewBlocks, 256, 0, stream>>>((float4*)conv, Cv, NH4);

        residual_kernel<<<resBlocks, 256, 0, stream>>>(conv, Wp(p + 2), bp(p + 2), conv, N);
        residual_kernel<<<resBlocks, 256, 0, stream>>>(conv, Wp(p + 3), bp(p + 3), conv, N);

        mul_kernel<<<ewBlocks, 256, 0, stream>>>((float4*)s_src, (const float4*)conv, NH4);

        residual_kernel<<<resBlocks, 256, 0, stream>>>(s_src, Wp(p + 4), bp(p + 4), s_src, N);
        residual_kernel<<<resBlocks, 256, 0, stream>>>(s_src, Wp(p + 5), bp(p + 5), s_src, N);

        add_kernel<<<ewBlocks, 256, 0, stream>>>((float4*)s_cur, (const float4*)s_src, NH4);
    }

    hipMemsetAsync(out, 0, (size_t)NGRAPHS * HIDDEN * sizeof(float), stream);
    pool_kernel<<<poolBlocks, 256, 0, stream>>>(s_cur, batch, out, N);
    for (int j = 0; j < 3; ++j)
        residual_kernel<<<1, 256, 0, stream>>>(out, Wp(12 + j), bp(12 + j), out, NGRAPHS);
}
